// SelfAttention_34591666602837
// MI455X (gfx1250) — matrix operand order repaired and checked by reading, not compiled, not run on hardware
//
#include <hip/hip_runtime.h>
#include <hip/hip_bf16.h>

#define B_    2
#define S_    2048
#define HID_  3584
#define NH_   28
#define NKV_  4
#define HD_   128
#define G_    7
#define NQ_   (NH_*HD_)        // 3584
#define NKVD_ (NKV_*HD_)       // 512
#define NTOT_ (NQ_+2*NKVD_)    // 4608
#define M_    (B_*S_)          // 4096

typedef __bf16 bf16;
typedef __attribute__((ext_vector_type(16))) __bf16 bf16x16;
typedef __attribute__((ext_vector_type(8)))  __bf16 bf16x8;
typedef __attribute__((ext_vector_type(8)))  float  f32x8;

#define WMMA_BF16(a,b,c) \
  __builtin_amdgcn_wmma_f32_16x16x32_bf16(false,(a),false,(b),(short)0,(c),false,false)

// LDS byte offset of a generic pointer to __shared__ data.
// Per CDNA5 ISA 10.2 the LDS offset is the low 32 bits of the address.
__device__ __forceinline__ uint32_t lds_off_of(const void* p) {
  return (uint32_t)(uintptr_t)p;
}

// Async global -> LDS 16B fill (GLOBAL_LOAD_ASYNC_TO_LDS_B128, ASYNCcnt).
// Per-lane: LDS[ldsoff] = MEM[gaddr]; no VGPR data, no staging registers.
__device__ __forceinline__ void async_ld_b128(uint32_t ldsoff, const void* g) {
  asm volatile("global_load_async_to_lds_b128 %0, %1, off"
               :: "v"(ldsoff), "v"((unsigned long long)(uintptr_t)g)
               : "memory");
}
__device__ __forceinline__ void wait_asynccnt0() {
  asm volatile("s_wait_asynccnt 0" ::: "memory");
}

// A fragment: 16x32 bf16 (MxK). Lane (h=lane>>4, m=lane&15) holds row m:
// elements 0..7 = K = 8h+{0..7}, elements 8..15 = K = 16+8h+{0..7}.
__device__ __forceinline__ bf16x16 load_a_frag(const bf16* base, int ld,
                                               int row0, int k0, int lane) {
  int h = lane >> 4, m = lane & 15;
  const bf16* p = base + (size_t)(row0 + m) * ld + k0 + h * 8;
  bf16x8 lo = *(const bf16x8*)(p);
  bf16x8 hi = *(const bf16x8*)(p + 16);
  bf16x16 r;
#pragma unroll
  for (int i = 0; i < 8; ++i) { r[i] = lo[i]; r[8 + i] = hi[i]; }
  return r;
}

// B fragment: 32x16 bf16 (KxN) from row-major W[N][K]. Lane (g=lane>>4,
// n=lane&15) holds column n: elements 0..7 = K = 8g+{0..7}, 8..15 = K = 16+8g+{0..7}.
__device__ __forceinline__ bf16x16 load_b_frag(const bf16* base, int ld,
                                               int n0, int k0, int lane) {
  int g = lane >> 4, n = lane & 15;
  const bf16* p = base + (size_t)(n0 + n) * ld + k0 + g * 8;
  bf16x8 lo = *(const bf16x8*)(p);
  bf16x8 hi = *(const bf16x8*)(p + 16);
  bf16x16 r;
#pragma unroll
  for (int i = 0; i < 8; ++i) { r[i] = lo[i]; r[8 + i] = hi[i]; }
  return r;
}

// ---------------------------------------------------------------------------
// bf16 GEMM, double-buffered LDS staging via async global->LDS DMA:
//   C[M,N] (f32) = A[M,K] @ W[N,K]^T (+ bias[N])
// Block = 256 thr = 8 waves (2 in M x 4 in N); block tile 128x256, BK=32;
// wave tile = 64x64 (4x4 WMMA). Per K-step the block stages 24KB into LDS
// once (async, no staging VGPRs), instead of 64KB of redundant L2 reads.
// Grid: (N/256, M/128).
// ---------------------------------------------------------------------------
__global__ __launch_bounds__(256) void gemm_bf16_kernel(
    const bf16* __restrict__ A, const bf16* __restrict__ W,
    const float* __restrict__ bias, float* __restrict__ C, int K, int ldc) {
  __shared__ bf16 Alds[2][128 * 32];   // 2 x 8KB
  __shared__ bf16 Blds[2][256 * 32];   // 2 x 16KB
  int tid  = threadIdx.x;
  int lane = tid & 31, wave = tid >> 5;
  int rowBlk = blockIdx.y * 128;
  int colBlk = blockIdx.x * 256;
  int wrow = (wave & 1) * 64;          // wave tile origin inside block tile
  int wcol = (wave >> 1) * 64;

  // Staging map: A -> thread t fills 32B (row=t>>1, kchunk=(t&1)*16)
  //              B -> thread t fills 64B (col=t, k 0..31)
  int arow = tid >> 1, achk = (tid & 1) * 16;
  const bf16* aptr = A + (size_t)(rowBlk + arow) * K + achk;
  const bf16* wptr = W + (size_t)(colBlk + tid) * K;
  uint32_t aoff[2] = { lds_off_of(&Alds[0][arow * 32 + achk]),
                       lds_off_of(&Alds[1][arow * 32 + achk]) };
  uint32_t boff[2] = { lds_off_of(&Blds[0][tid * 32]),
                       lds_off_of(&Blds[1][tid * 32]) };

  auto stage = [&](int bufi, int k0) {
    const bf16* ap = aptr + k0;
    const bf16* wp = wptr + k0;
    async_ld_b128(aoff[bufi],      ap);
    async_ld_b128(aoff[bufi] + 16, ap + 8);
    async_ld_b128(boff[bufi],      wp);
    async_ld_b128(boff[bufi] + 16, wp + 8);
    async_ld_b128(boff[bufi] + 32, wp + 16);
    async_ld_b128(boff[bufi] + 48, wp + 24);
  };

  stage(0, 0);
  f32x8 acc[4][4] = {};
  int buf = 0;
  for (int k0 = 0; k0 < K; k0 += 32) {
    wait_asynccnt0();                  // my fills of `buf` are in LDS
    __syncthreads();                   // everyone's fills visible
    if (k0 + 32 < K) stage(buf ^ 1, k0 + 32);   // overlap next slab with math
    bf16x16 a[4];
#pragma unroll
    for (int mt = 0; mt < 4; ++mt)
      a[mt] = load_a_frag(&Alds[buf][0], 32, wrow + mt*16, 0, lane);
#pragma unroll
    for (int nt = 0; nt < 4; ++nt) {
      bf16x16 w = load_b_frag(&Blds[buf][0], 32, wcol + nt*16, 0, lane);
#pragma unroll
      for (int mt = 0; mt < 4; ++mt)
        acc[mt][nt] = WMMA_BF16(a[mt], w, acc[mt][nt]);
    }
    buf ^= 1;
  }

  // C layout: element r -> row (r + 8*(lane>>4)), col (lane&15) within tile.
  int lhalf = lane >> 4, n = lane & 15;
#pragma unroll
  for (int mt = 0; mt < 4; ++mt)
#pragma unroll
    for (int nt = 0; nt < 4; ++nt)
#pragma unroll
      for (int r = 0; r < 8; ++r) {
        int row = rowBlk + wrow + mt*16 + r + 8*lhalf;
        int col = colBlk + wcol + nt*16 + n;
        float v = acc[mt][nt][r];
        if (bias) v += bias[col];
        C[(size_t)row * ldc + col] = v;
      }
}

// ---------------------------------------------------------------------------
__global__ void cvt_kernel(const float* __restrict__ src, bf16* __restrict__ dst, int n) {
  int i = blockIdx.x * blockDim.x + threadIdx.x;
  if (i < n) dst[i] = (bf16)src[i];
}

__global__ void pack_bias_kernel(const float* __restrict__ bq, const float* __restrict__ bk,
                                 const float* __restrict__ bv, float* __restrict__ bias) {
  int i = blockIdx.x * blockDim.x + threadIdx.x;
  if (i < NQ_)            bias[i] = bq[i];
  else if (i < NQ_+NKVD_) bias[i] = bk[i - NQ_];
  else if (i < NTOT_)     bias[i] = bv[i - NQ_ - NKVD_];
}

// ---------------------------------------------------------------------------
// RoPE + split + convert. qkv f32 [M, 4608] ->
//   qb  bf16 [b][s][NQ]   (RoPE'd, pre-scaled by log2(e)/sqrt(HD) so softmax
//                          runs in base-2: p = exp2(s - m))
//   kb  bf16 [b][s][NKVD] (RoPE'd)
//   vT  bf16 [b][h][hd][s] (transposed for PV B-fragments)
// ---------------------------------------------------------------------------
__global__ __launch_bounds__(256) void rope_kernel(
    const float* __restrict__ qkv, const float* __restrict__ ct,
    const float* __restrict__ st, bf16* __restrict__ qb,
    bf16* __restrict__ kb, bf16* __restrict__ vT) {
  int col = blockIdx.x * blockDim.x + threadIdx.x;   // 0..4607
  int row = blockIdx.y;                              // 0..M-1
  int s = row & (S_ - 1);
  int b = row >> 11;                                 // S_ == 2048
  float x = qkv[(size_t)row * NTOT_ + col];
  if (col < NQ_) {
    int d = col & (HD_ - 1);
    float c = ct[s*HD_ + d], sn = st[s*HD_ + d];
    float other = qkv[(size_t)row * NTOT_ + (d < 64 ? col + 64 : col - 64)];
    float r = (d < 64) ? (x*c - other*sn) : (x*c + other*sn);
    qb[(size_t)row * NQ_ + col] = (bf16)(r * 0.12751743f); // log2(e)/sqrt(128)
  } else if (col < NQ_ + NKVD_) {
    int kc = col - NQ_;
    int d = kc & (HD_ - 1);
    float c = ct[s*HD_ + d], sn = st[s*HD_ + d];
    float other = qkv[(size_t)row * NTOT_ + (d < 64 ? col + 64 : col - 64)];
    float r = (d < 64) ? (x*c - other*sn) : (x*c + other*sn);
    kb[(size_t)row * NKVD_ + kc] = (bf16)r;
  } else {
    int vc = col - (NQ_ + NKVD_);
    int hh = vc >> 7, d = vc & 127;
    vT[((size_t)(b*NKV_ + hh) * HD_ + d) * S_ + s] = (bf16)x;
  }
}

// ---------------------------------------------------------------------------
// One 64-key flash-attention block for one wave (16 q-rows x HD=128).
// MASK=false for blocks fully below the causal diagonal (no compares).
// 16 WMMAs QK^T + 16 WMMAs PV; P relaid C->A layout via 2KB/wave LDS tile.
// ---------------------------------------------------------------------------
template <bool MASK>
__device__ __forceinline__ void attn_block(
    int kblk, int q0, int lane, int lhalf, int n,
    const bf16* kbase, const bf16* vbase, bf16* pl,
    const bf16x16 (&qa)[4], f32x8 (&o)[8], float (&m_r)[8], float (&l_r)[8]) {
  f32x8 s[4] = {};
#pragma unroll
  for (int ki = 0; ki < 4; ++ki)
#pragma unroll
    for (int nt = 0; nt < 4; ++nt) {
      bf16x16 bk = load_b_frag(kbase, NKVD_, kblk + nt*16, ki*32, lane);
      s[nt] = WMMA_BF16(qa[ki], bk, s[nt]);
    }
#pragma unroll
  for (int r = 0; r < 8; ++r) {
    int row = q0 + r + 8*lhalf;
    float v[4];
#pragma unroll
    for (int nt = 0; nt < 4; ++nt) {
      v[nt] = s[nt][r];
      if (MASK) v[nt] = (kblk + nt*16 + n <= row) ? v[nt] : -1e30f;
    }
    float mx = fmaxf(fmaxf(v[0], v[1]), fmaxf(v[2], v[3]));
#pragma unroll
    for (int off = 1; off < 16; off <<= 1) mx = fmaxf(mx, __shfl_xor(mx, off, 32));
    float mnew  = fmaxf(m_r[r], mx);
    float alpha = exp2f(m_r[r] - mnew);
    float rs = 0.f;
#pragma unroll
    for (int nt = 0; nt < 4; ++nt) {
      float p = exp2f(v[nt] - mnew);
      rs += p;
      pl[(r + 8*lhalf)*64 + nt*16 + n] = (bf16)p;   // P -> LDS, row-major 16x64
    }
#pragma unroll
    for (int off = 1; off < 16; off <<= 1) rs += __shfl_xor(rs, off, 32);
    l_r[r] = l_r[r] * alpha + rs;
    m_r[r] = mnew;
#pragma unroll
    for (int j = 0; j < 8; ++j) o[j][r] *= alpha;
  }
  asm volatile("s_wait_dscnt 0" ::: "memory");      // wave-internal LDS RAW
#pragma unroll
  for (int kk = 0; kk < 2; ++kk) {                  // P as two 16x32 A-frags
    bf16x16 pa;
    const bf16* pp = pl + n*64 + kk*32 + lhalf*8;   // A-frag row = lane&15
    bf16x8 lo = *(const bf16x8*)pp;
    bf16x8 hi = *(const bf16x8*)(pp + 16);
#pragma unroll
    for (int i = 0; i < 8; ++i) { pa[i] = lo[i]; pa[8+i] = hi[i]; }
#pragma unroll
    for (int j = 0; j < 8; ++j) {
      bf16x16 bv = load_b_frag(vbase, S_, j*16, kblk + kk*32, lane);
      o[j] = WMMA_BF16(pa, bv, o[j]);
    }
  }
}

// Flash attention (causal, GQA). Block = 256 thr = 8 waves; each wave owns
// 16 q-rows x full HD=128. Grid: (S/128, NH, B).
__global__ __launch_bounds__(256) void attn_kernel(
    const bf16* __restrict__ qb, const bf16* __restrict__ kb,
    const bf16* __restrict__ vT, bf16* __restrict__ ab) {
  __shared__ bf16 plds[8][16 * 64];                 // 2KB per wave
  int lane = threadIdx.x & 31, wave = threadIdx.x >> 5;
  int q0 = blockIdx.x * 128 + wave * 16;
  int hq = blockIdx.y;
  int b  = blockIdx.z;
  int h  = hq / G_;                                 // KV head for this Q head

  const bf16* qbase = qb + (size_t)b * S_ * NQ_ + hq * HD_;
  const bf16* kbase = kb + (size_t)b * S_ * NKVD_ + h * HD_;
  const bf16* vbase = vT + ((size_t)(b * NKV_ + h)) * HD_ * S_;

  bf16x16 qa[4];
#pragma unroll
  for (int ki = 0; ki < 4; ++ki) qa[ki] = load_a_frag(qbase, NQ_, q0, ki*32, lane);

  f32x8 o[8] = {};
  float m_r[8], l_r[8];
#pragma unroll
  for (int r = 0; r < 8; ++r) { m_r[r] = -1e30f; l_r[r] = 0.f; }

  int lhalf = lane >> 4, n = lane & 15;
  bf16* pl = &plds[wave][0];

  int kend = q0 & ~63;                              // full (unmasked) region
  for (int kblk = 0; kblk < kend; kblk += 64)
    attn_block<false>(kblk, q0, lane, lhalf, n, kbase, vbase, pl, qa, o, m_r, l_r);
  attn_block<true>(kend, q0, lane, lhalf, n, kbase, vbase, pl, qa, o, m_r, l_r);

#pragma unroll
  for (int r = 0; r < 8; ++r) {
    float inv = 1.0f / l_r[r];
    int row = q0 + r + 8*lhalf;
    bf16* orow = ab + (size_t)(b * S_ + row) * NQ_ + hq * HD_;
#pragma unroll
    for (int j = 0; j < 8; ++j) orow[j*16 + n] = (bf16)(o[j][r] * inv);
  }
}

// ---------------------------------------------------------------------------
extern "C" void kernel_launch(void* const* d_in, const int* in_sizes, int n_in,
                              void* d_out, int out_size, void* d_ws, size_t ws_size,
                              hipStream_t stream) {
  const float* hs = (const float*)d_in[0];
  const float* wq = (const float*)d_in[1];
  const float* bq = (const float*)d_in[2];
  const float* wk = (const float*)d_in[3];
  const float* bk = (const float*)d_in[4];
  const float* wv = (const float*)d_in[5];
  const float* bv = (const float*)d_in[6];
  const float* wo = (const float*)d_in[7];
  const float* ct = (const float*)d_in[8];
  const float* st = (const float*)d_in[9];

  char* ws = (char*)d_ws;
  size_t off = 0;
  auto alloc = [&](size_t bytes) {
    char* p = ws + off; off += (bytes + 255) & ~(size_t)255; return p;
  };
  bf16*  hsb  = (bf16*) alloc((size_t)M_   * HID_ * 2);
  bf16*  wb   = (bf16*) alloc((size_t)NTOT_* HID_ * 2);   // wq|wk|wv stacked
  bf16*  wob  = (bf16*) alloc((size_t)HID_ * NQ_  * 2);
  bf16*  qbuf = (bf16*) alloc((size_t)M_   * NQ_  * 2);
  bf16*  kbuf = (bf16*) alloc((size_t)M_   * NKVD_* 2);
  bf16*  vT   = (bf16*) alloc((size_t)B_ * NKV_ * HD_ * S_ * 2);
  bf16*  ab   = (bf16*) alloc((size_t)M_   * NQ_  * 2);
  float* bias = (float*)alloc((size_t)NTOT_ * 4);
  float* qkvf = (float*)alloc((size_t)M_ * NTOT_ * 4);

  auto cvtl = [&](const float* s, bf16* d, int n) {
    cvt_kernel<<<(n + 255)/256, 256, 0, stream>>>(s, d, n);
  };
  cvtl(hs, hsb, M_ * HID_);
  cvtl(wq, wb,                               NQ_   * HID_);
  cvtl(wk, wb + (size_t)NQ_ * HID_,          NKVD_ * HID_);
  cvtl(wv, wb + (size_t)(NQ_+NKVD_) * HID_,  NKVD_ * HID_);
  cvtl(wo, wob, HID_ * NQ_);
  pack_bias_kernel<<<(NTOT_ + 255)/256, 256, 0, stream>>>(bq, bk, bv, bias);

  // QKV projection: [M,4608] = hs @ [wq|wk|wv]^T + bias
  gemm_bf16_kernel<<<dim3(NTOT_/256, M_/128), 256, 0, stream>>>(
      hsb, wb, bias, qkvf, HID_, NTOT_);
  // RoPE + split + V transpose (+ log2(e)/sqrt(HD) folded into Q)
  rope_kernel<<<dim3(NTOT_/256, M_), 256, 0, stream>>>(qkvf, ct, st, qbuf, kbuf, vT);
  // Flash attention
  attn_kernel<<<dim3(S_/128, NH_, B_), 256, 0, stream>>>(qbuf, kbuf, vT, ab);
  // Output projection: out[M,HID] = ab @ wo^T
  gemm_bf16_kernel<<<dim3(HID_/256, M_/128), 256, 0, stream>>>(
      ab, wob, nullptr, (float*)d_out, NQ_, HID_);
}
